// Attention_11398843203859
// MI455X (gfx1250) — matrix operand order repaired and checked by reading, not compiled, not run on hardware
//
#include <hip/hip_runtime.h>
#include <hip/hip_bf16.h>
#include <math.h>

// ---------------------------------------------------------------------------
// Bahdanau attention, fused for MI455X (gfx1250, wave32, WMMA).
//   B=32, S=4096, HIDDEN=ANN=512
//   score[b,s] = w2 . tanh(preb[b,:] + ann[b,s,:] @ W1_a)   (b1 folded into preb,
//                                                            b2 drops out of softmax)
//   alpha = softmax_s(score);  context[b,:] = alpha @ ann[b]
// Main GEMM: bf16 WMMA 16x16x32 (f32 accum). Each wave owns a 32x64 register
// tile (2 M-subtiles x 4 N-subtiles) so each B fragment is reused x2 and each
// A fragment x4 from registers. Annotations (268 MB, > L2) are streamed with
// non-temporal loads; W1^T (512 KB bf16) stays hot in L2. tanh uses the CDNA5
// hardware transcendental (v_tanh_f32) when the toolchain exposes it.
// ---------------------------------------------------------------------------

#define HIDDEN 512
#define ANN    512
#define BATCH  32
#define SEQ    4096
#define SROWS  256          // S-rows per workgroup (8 waves x 32 rows)
#define LDST   520          // padded bf16 row stride in LDS (spreads banks)

typedef __attribute__((ext_vector_type(16))) __bf16 v16bf;
typedef __attribute__((ext_vector_type(8)))  __bf16 v8bf;
typedef __attribute__((ext_vector_type(8)))  float  v8f;
typedef __attribute__((ext_vector_type(4)))  float  v4f;   // clang-native float4

#if defined(__has_builtin)
#if __has_builtin(__builtin_amdgcn_tanhf)
#define FAST_TANH(x) __builtin_amdgcn_tanhf(x)
#endif
#endif
#ifndef FAST_TANH
#define FAST_TANH(x) tanhf(x)
#endif

// ---------------- kernel 0a: preb[b,n] = b1[n] + sum_k ph[b,k]*W1[k,n] -------
__global__ __launch_bounds__(256) void preb_kernel(
    const float* __restrict__ ph, const float* __restrict__ W1,
    const float* __restrict__ b1, float* __restrict__ preb) {
  __shared__ float h[HIDDEN];
  const int b = blockIdx.x, tid = threadIdx.x;
  for (int i = tid; i < HIDDEN; i += 256) h[i] = ph[b * HIDDEN + i];
  __syncthreads();
  #pragma unroll
  for (int rep = 0; rep < 2; ++rep) {
    const int n = tid + rep * 256;
    float acc = b1[n];
    for (int k = 0; k < HIDDEN; ++k) acc += h[k] * W1[(size_t)k * HIDDEN + n];
    preb[b * HIDDEN + n] = acc;
  }
}

// ------- kernel 0b: W1_a (f32, [k][n]) -> bf16 transposed [n][k] -------------
__global__ __launch_bounds__(256) void w1a_convert_kernel(
    const float* __restrict__ W1, __bf16* __restrict__ w1at) {
  __shared__ float t[32][33];
  const int k0 = blockIdx.x * 32, n0 = blockIdx.y * 32;
  const int kk = threadIdx.x >> 5, nn = threadIdx.x & 31;
  #pragma unroll
  for (int r = 0; r < 4; ++r) {
    const int k = k0 + kk + 8 * r;
    t[kk + 8 * r][nn] = W1[(size_t)(HIDDEN + k) * HIDDEN + n0 + nn];
  }
  __syncthreads();
  #pragma unroll
  for (int r = 0; r < 4; ++r) {
    const int n = n0 + kk + 8 * r;
    w1at[(size_t)n * ANN + k0 + nn] = (__bf16)t[nn][kk + 8 * r];
  }
}

// ---------------- kernel 1: fused GEMM + tanh + dot(w2) ----------------------
// grid (SEQ/SROWS=16, BATCH=32), 256 threads = 8 waves. Dynamic LDS:
//   As:   SROWS * LDST bf16  (annotation tile, bf16)   ~260 KB
//   scl:  SROWS floats       (per-row score partials via ds_add_f32)
__global__ __launch_bounds__(256) void score_kernel(
    const float* __restrict__ ann, const __bf16* __restrict__ w1at,
    const float* __restrict__ preb, const float* __restrict__ W2,
    float* __restrict__ scores) {
  extern __shared__ char smem[];
  __bf16* As = (__bf16*)smem;
  float* scl = (float*)(smem + (size_t)SROWS * LDST * sizeof(__bf16));

  const int tid   = threadIdx.x;
  const int b     = blockIdx.y;
  const int sbase = blockIdx.x * SROWS;
  const int lane  = tid & 31;
  const int wv    = tid >> 5;        // wave id 0..7
  const int lmod  = lane & 15;
  const int khalf = lane >> 4;       // 0/1: which K half this lane holds

  if (tid < SROWS) scl[tid] = 0.0f;

  // Stage 256x512 f32 annotations -> bf16 LDS. Non-temporal: one-shot stream,
  // keep it out of L2 (which holds the shared W1^T).
  const float* atile = ann + ((size_t)b * SEQ + sbase) * ANN;
  for (int i = tid; i < SROWS * (ANN / 4); i += 256) {
    const int row = i >> 7;          // ANN/4 = 128 float4 per row
    const int c4  = i & 127;
    const v4f v =
        __builtin_nontemporal_load((const v4f*)(atile + (size_t)row * ANN) + c4);
    union { __bf16 h[4]; unsigned long long u; } pk;
    pk.h[0] = (__bf16)v.x; pk.h[1] = (__bf16)v.y;
    pk.h[2] = (__bf16)v.z; pk.h[3] = (__bf16)v.w;
    *(unsigned long long*)(As + (size_t)row * LDST + c4 * 4) = pk.u;
  }
  __syncthreads();

  // Per-lane A fragment rows (16-bit 16x32 A layout):
  //   lane<16 : K in {k0..k0+7, k0+16..k0+23};  lane>=16: shifted by 8.
  // This wave owns M rows [wv*32, wv*32+32): two 16-row subtiles.
  const __bf16* Arow0 = As + (size_t)(wv * 32 + lmod) * LDST;
  const __bf16* Arow1 = Arow0 + (size_t)16 * LDST;
  const float* prebb = preb + (size_t)b * HIDDEN;

  float sacc[2][8];
  #pragma unroll
  for (int m = 0; m < 2; ++m)
    #pragma unroll
    for (int v = 0; v < 8; ++v) sacc[m][v] = 0.0f;

  for (int c = 0; c < HIDDEN / 64; ++c) {          // 8 N-chunks of 64
    v8f acc[2][4];
    #pragma unroll
    for (int m = 0; m < 2; ++m)
      #pragma unroll
      for (int j = 0; j < 4; ++j) acc[m][j] = (v8f)0.0f;

    #pragma unroll 4
    for (int k0 = 0; k0 < ANN; k0 += 32) {         // 16 K-steps
      const v8bf a0l = *(const v8bf*)(Arow0 + k0 + khalf * 8);
      const v8bf a0h = *(const v8bf*)(Arow0 + k0 + 16 + khalf * 8);
      const v16bf a0 = __builtin_shufflevector(a0l, a0h, 0, 1, 2, 3, 4, 5, 6, 7,
                                               8, 9, 10, 11, 12, 13, 14, 15);
      const v8bf a1l = *(const v8bf*)(Arow1 + k0 + khalf * 8);
      const v8bf a1h = *(const v8bf*)(Arow1 + k0 + 16 + khalf * 8);
      const v16bf a1 = __builtin_shufflevector(a1l, a1h, 0, 1, 2, 3, 4, 5, 6, 7,
                                               8, 9, 10, 11, 12, 13, 14, 15);
      #pragma unroll
      for (int j = 0; j < 4; ++j) {
        // B layout: lane n holds K {0..7, 16..23} of column n; lane n+16 shifted by 8.
        const int ncol = c * 64 + j * 16 + lmod;
        const v8bf bl = *(const v8bf*)(w1at + (size_t)ncol * ANN + k0 + khalf * 8);
        const v8bf bh = *(const v8bf*)(w1at + (size_t)ncol * ANN + k0 + 16 + khalf * 8);
        const v16bf bf = __builtin_shufflevector(bl, bh, 0, 1, 2, 3, 4, 5, 6, 7,
                                                 8, 9, 10, 11, 12, 13, 14, 15);
        acc[0][j] = __builtin_amdgcn_wmma_f32_16x16x32_bf16(
            false, a0, false, bf, (short)0, acc[0][j], false, false);
        acc[1][j] = __builtin_amdgcn_wmma_f32_16x16x32_bf16(
            false, a1, false, bf, (short)0, acc[1][j], false, false);
      }
    }

    // Fold tanh + w2 dot. C element (vgpr v, lane): M = 32*wv + m*16 + v + 8*khalf,
    // N = c*64 + j*16 + lmod. preb/w2 depend only on N (same for all m, v).
    #pragma unroll
    for (int j = 0; j < 4; ++j) {
      const int ng = c * 64 + j * 16 + lmod;
      const float p = prebb[ng];
      const float w = W2[ng];
      #pragma unroll
      for (int m = 0; m < 2; ++m)
        #pragma unroll
        for (int v = 0; v < 8; ++v)
          sacc[m][v] += FAST_TANH(p + acc[m][j][v]) * w;
    }
  }

  // 16 lanes per row-half contribute to the same M: reduce via LDS f32 atomics.
  #pragma unroll
  for (int m = 0; m < 2; ++m) {
    const int mbase = wv * 32 + m * 16 + khalf * 8;
    #pragma unroll
    for (int v = 0; v < 8; ++v) atomicAdd(&scl[mbase + v], sacc[m][v]);
  }
  __syncthreads();
  if (tid < SROWS) scores[(size_t)b * SEQ + sbase + tid] = scl[tid];
}

// ---------------- kernel 2: per-batch softmax over S (in place) --------------
__global__ __launch_bounds__(256) void softmax_kernel(float* __restrict__ scores) {
  __shared__ float sh[SEQ];
  __shared__ float red[256];
  const int b = blockIdx.x, tid = threadIdx.x;
  float m = -3.4e38f;
  for (int i = tid; i < SEQ; i += 256) {
    const float v = scores[(size_t)b * SEQ + i];
    sh[i] = v;
    m = fmaxf(m, v);
  }
  red[tid] = m; __syncthreads();
  for (int s = 128; s > 0; s >>= 1) {
    if (tid < s) red[tid] = fmaxf(red[tid], red[tid + s]);
    __syncthreads();
  }
  const float mx = red[0]; __syncthreads();
  float sum = 0.0f;
  for (int i = tid; i < SEQ; i += 256) {
    const float e = expf(sh[i] - mx);
    sh[i] = e;
    sum += e;
  }
  red[tid] = sum; __syncthreads();
  for (int s = 128; s > 0; s >>= 1) {
    if (tid < s) red[tid] += red[tid + s];
    __syncthreads();
  }
  const float inv = 1.0f / red[0];
  for (int i = tid; i < SEQ; i += 256)
    scores[(size_t)b * SEQ + i] = sh[i] * inv;
}

// ---------------- kernel 3: context[b,a] = sum_s alpha[b,s]*ann[b,s,a] -------
// grid (ANN/256=2, BATCH), 256 threads; coalesced non-temporal streaming pass.
__global__ __launch_bounds__(256) void context_kernel(
    const float* __restrict__ ann, const float* __restrict__ alpha,
    float* __restrict__ out) {
  __shared__ float al[SEQ];
  const int b = blockIdx.y, a0 = blockIdx.x * 256, tid = threadIdx.x;
  for (int i = tid; i < SEQ; i += 256) al[i] = alpha[(size_t)b * SEQ + i];
  __syncthreads();
  const float* base = ann + (size_t)b * SEQ * ANN + a0 + tid;
  float acc = 0.0f;
  #pragma unroll 8
  for (int s = 0; s < SEQ; ++s)
    acc += al[s] * __builtin_nontemporal_load(base + (size_t)s * ANN);
  out[(size_t)b * ANN + a0 + tid] = acc;
}

// ---------------------------------------------------------------------------
extern "C" void kernel_launch(void* const* d_in, const int* in_sizes, int n_in,
                              void* d_out, int out_size, void* d_ws, size_t ws_size,
                              hipStream_t stream) {
  const float* ph  = (const float*)d_in[0];   // (32, 512)
  const float* ann = (const float*)d_in[1];   // (32, 4096, 512)
  const float* W1  = (const float*)d_in[2];   // (1024, 512)
  const float* b1  = (const float*)d_in[3];   // (512,)
  const float* W2  = (const float*)d_in[4];   // (512, 1)
  // d_in[5] = b2: drops out of softmax, unused.
  float* out = (float*)d_out;                 // (32, 1, 512)

  char* ws = (char*)d_ws;
  float*  preb   = (float*)ws;                                  //  64 KB
  __bf16* w1at   = (__bf16*)(ws + (size_t)BATCH * HIDDEN * 4);  // 512 KB
  float*  scores = (float*)(ws + (size_t)BATCH * HIDDEN * 4
                               + (size_t)HIDDEN * ANN * 2);     // 512 KB

  preb_kernel<<<BATCH, 256, 0, stream>>>(ph, W1, b1, preb);
  w1a_convert_kernel<<<dim3(ANN / 32, HIDDEN / 32), 256, 0, stream>>>(W1, w1at);

  const size_t score_lds = (size_t)SROWS * LDST * sizeof(__bf16)
                         + (size_t)SROWS * sizeof(float);       // ~261 KB of 320 KB
  score_kernel<<<dim3(SEQ / SROWS, BATCH), 256, score_lds, stream>>>(
      ann, w1at, preb, W2, scores);

  softmax_kernel<<<BATCH, 256, 0, stream>>>(scores);
  context_kernel<<<dim3(ANN / 256, BATCH), 256, 0, stream>>>(ann, scores, out);
  (void)in_sizes; (void)n_in; (void)out_size; (void)ws_size;
}
